// DiagonalSSM_38766374813855
// MI455X (gfx1250) — matrix operand order repaired and checked by reading, not compiled, not run on hardware
//
#include <hip/hip_runtime.h>
#include <hip/hip_bf16.h>

typedef __attribute__((ext_vector_type(16))) _Float16 v16h;
typedef __attribute__((ext_vector_type(8)))  float    v8f;

#define BATCH  32
#define SEQLEN 2048
#define NSTATE 2048
#define PLEN   (2 * SEQLEN)   // reversed kernel + zero pad (f16)

// ---------------------------------------------------------------------------
// Kernel 1: build reversed+padded Toeplitz kernel P in f16:
//   P[SEQLEN-1-k] = S(k) = sum_n A_diag[n]^k   (k = 0..SEQLEN-1)
//   P[SEQLEN .. 2*SEQLEN) = 0                  (encodes causal mask)
// 1024 threads; running powers; wave32 shuffle butterfly + 32-partial LDS
// stage (2 barriers per k instead of 11).
// ---------------------------------------------------------------------------
__global__ void __launch_bounds__(1024)
ssm_powsum_kernel(const float* __restrict__ A_diag, _Float16* __restrict__ P) {
    __shared__ float red[32];
    const int tid  = threadIdx.x;
    const int lane = tid & 31;
    const int wid  = tid >> 5;

    // zero pad region (causal mask lives here)
    P[SEQLEN + tid]        = (_Float16)0.0f;
    P[SEQLEN + 1024 + tid] = (_Float16)0.0f;

    const float a0 = A_diag[tid];
    const float a1 = A_diag[tid + 1024];
    float p0 = 1.0f, p1 = 1.0f;

    for (int k = 0; k < SEQLEN; ++k) {
        // wave32 butterfly reduction of the 2-way partial
        float v = p0 + p1;
        #pragma unroll
        for (int m = 16; m > 0; m >>= 1)
            v += __shfl_xor(v, m, 32);
        if (lane == 0) red[wid] = v;
        __syncthreads();
        if (wid == 0) {
            float w = red[lane];
            #pragma unroll
            for (int m = 16; m > 0; m >>= 1)
                w += __shfl_xor(w, m, 32);
            if (lane == 0) P[SEQLEN - 1 - k] = (_Float16)w;
        }
        p0 *= a0;
        p1 *= a1;
        __syncthreads();   // protect red[] before next iteration overwrites
    }
}

// ---------------------------------------------------------------------------
// Kernel 2: Y[b,t] = alpha * sum_{s<=t} S(t-s) * x[b,s]
//         = causal Toeplitz GEMM [32 x 2048] * [2048 x 2048], WMMA f16->f32.
// B[s,t] = P[(SEQLEN-1-t) + s]  -- branch-free via the zero pad.
// P staged in LDS via ASYNC global->LDS b128 copies (no VGPR bounce).
// One wave per 16-wide time-tile; BOTH batch tiles per wave (B reused
// across two WMMAs). 128 waves total, 4 waves/block, 32 blocks.
// ---------------------------------------------------------------------------
__global__ void __launch_bounds__(128)
ssm_toeplitz_wmma_kernel(const float* __restrict__ x,
                         const _Float16* __restrict__ P,
                         const float* __restrict__ alpha_p,
                         float* __restrict__ y) {
    __shared__ _Float16 Pl[PLEN];   // 8 KB

    // --- CDNA5 async copy: global -> LDS, 128 lanes x 16B x 4 passes ---
    {
        unsigned           ldsbase = (unsigned)(uintptr_t)Pl + threadIdx.x * 16u;
        unsigned long long gbase   = (unsigned long long)(uintptr_t)P + threadIdx.x * 16u;
        #pragma unroll
        for (int i = 0; i < 4; ++i) {
            unsigned           l = ldsbase + (unsigned)(i * 2048);
            unsigned long long g = gbase + (unsigned long long)(i * 2048);
            asm volatile("global_load_async_to_lds_b128 %0, %1, off"
                         :: "v"(l), "v"(g) : "memory");
        }
        asm volatile("s_wait_asynccnt 0" ::: "memory");
    }
    __syncthreads();

    const int wave = threadIdx.x >> 5;
    const int lane = threadIdx.x & 31;
    const int half = lane >> 4;           // lane 0-15 vs 16-31 (K-half select)
    const int l16  = lane & 15;

    const int nTile = blockIdx.x * 4 + wave;   // 0..127 time tiles
    const int ncol  = nTile * 16 + l16;        // B/D column (time index t)
    const int prow  = (SEQLEN - 1) - ncol;     // base into reversed kernel

    v8f acc0 = {};   // batch rows 0..15
    v8f acc1 = {};   // batch rows 16..31

    // Causality: B tile identically zero once kk > nTile*16+15, so bound K.
    int kend = (nTile * 16 + 16 + 31) & ~31;
    if (kend > SEQLEN) kend = SEQLEN;

    const float* xrow0 = x + (l16)      * SEQLEN;   // A row, batch tile 0
    const float* xrow1 = x + (16 + l16) * SEQLEN;   // A row, batch tile 1

    for (int kk = 0; kk < kend; kk += 32) {
        // --- A fragments: 16x32 f16 slices of x, contiguous k order ---
        // lane half h: VGPR v -> K=16h+2v,16h+2v+1 (contiguous, matches B)
        v16h a0, a1;
        #pragma unroll
        for (int v = 0; v < 8; ++v) {
            const int k0 = kk + 2 * v + 16 * half;
            a0[2 * v]     = (_Float16)xrow0[k0];
            a0[2 * v + 1] = (_Float16)xrow0[k0 + 1];
            a1[2 * v]     = (_Float16)xrow1[k0];
            a1[2 * v + 1] = (_Float16)xrow1[k0 + 1];
        }

        // --- B fragment (shared by both batch tiles): branch-free gather ---
        // N = lane&15 ; element e -> s = kk + e + 16*half ; B = Pl[prow + s]
        const int sbase = prow + kk + 16 * half;
        v16h b;
        #pragma unroll
        for (int e = 0; e < 16; ++e)
            b[e] = Pl[sbase + e];

        acc0 = __builtin_amdgcn_wmma_f32_16x16x32_f16(
            false, a0, false, b, (short)0, acc0, false, false);
        acc1 = __builtin_amdgcn_wmma_f32_16x16x32_f16(
            false, a1, false, b, (short)0, acc1, false, false);
    }

    // --- Store D: 32-bit C/D layout: VGPR v holds row M = v + 8*half ---
    const float alpha = alpha_p[0];
    #pragma unroll
    for (int v = 0; v < 8; ++v) {
        const int m = v + 8 * half;
        y[(m)      * SEQLEN + ncol] = alpha * acc0[v];
        y[(m + 16) * SEQLEN + ncol] = alpha * acc1[v];
    }
}

// ---------------------------------------------------------------------------
// Launch
// ---------------------------------------------------------------------------
extern "C" void kernel_launch(void* const* d_in, const int* in_sizes, int n_in,
                              void* d_out, int out_size, void* d_ws, size_t ws_size,
                              hipStream_t stream) {
    const float* x       = (const float*)d_in[0];   // [32, 2048] f32
    const float* A_diag  = (const float*)d_in[1];   // [2048]     f32
    const float* alpha_p = (const float*)d_in[2];   // scalar     f32
    float*       y       = (float*)d_out;           // [32, 2048] f32
    _Float16*    P       = (_Float16*)d_ws;         // 4096 f16 = 8 KB scratch

    // Stage 1: reversed+padded power-sum kernel P.
    ssm_powsum_kernel<<<1, 1024, 0, stream>>>(A_diag, P);

    // Stage 2: causal Toeplitz GEMM on the matrix pipe.
    // 128 time tiles, 1 wave each (both batch tiles), 4 waves per block.
    ssm_toeplitz_wmma_kernel<<<32, 128, 0, stream>>>(x, P, alpha_p, y);
}
